// LoRAAttention_37271726194873
// MI455X (gfx1250) — matrix operand order repaired and checked by reading, not compiled, not run on hardware
//
#include <hip/hip_runtime.h>

typedef float  v2f  __attribute__((ext_vector_type(2)));
typedef float  v4f  __attribute__((ext_vector_type(4)));
typedef float  v8f  __attribute__((ext_vector_type(8)));
typedef __bf16 v4bf __attribute__((ext_vector_type(4)));
typedef __bf16 v8bf __attribute__((ext_vector_type(8)));
typedef __bf16 v16bf __attribute__((ext_vector_type(16)));
typedef unsigned int u32x4 __attribute__((ext_vector_type(4)));
typedef int i32x4 __attribute__((ext_vector_type(4)));
typedef int i32x8 __attribute__((ext_vector_type(8)));

#define D_MODEL 4096
#define RANK    16
#define M_TOTAL (4 * 4096)

#define BM 128
#define BN 128
#define BK 64
#define PITCH 72   // bf16 units; 144B rows; matches TDM pad: 32 dwords data + 4 dwords pad

static __device__ inline v16bf cat8(v8bf lo, v8bf hi) {
    return __builtin_shufflevector(lo, hi, 0, 1, 2, 3, 4, 5, 6, 7,
                                   8, 9, 10, 11, 12, 13, 14, 15);
}

// -------------------------------------------------------------------------
// Kernel 0: split W (fp32) into W_hi + W_lo (bf16 each), one shot.
// -------------------------------------------------------------------------
__global__ __launch_bounds__(256) void split_w_kernel(const float* __restrict__ W,
                                                      __bf16* __restrict__ Wh,
                                                      __bf16* __restrict__ Wl) {
    const size_t i = ((size_t)blockIdx.x * 256 + threadIdx.x) * 4;
    v4f w = *(const v4f*)(W + i);
    v4bf h, l;
#pragma unroll
    for (int j = 0; j < 4; ++j) {
        __bf16 hj = (__bf16)w[j];
        h[j] = hj;
        l[j] = (__bf16)(w[j] - (float)hj);
    }
    *(v4bf*)(Wh + i) = h;
    *(v4bf*)(Wl + i) = l;
}

// -------------------------------------------------------------------------
// Kernel 1: u = x @ B^T  -> [M_TOTAL, RANK]   (exact fp32 WMMA path)
// -------------------------------------------------------------------------
__global__ __launch_bounds__(256) void lora_u_kernel(const float* __restrict__ x,
                                                     const float* __restrict__ Blora,
                                                     float* __restrict__ u) {
    const int lane = threadIdx.x & 31;
    const int wave = threadIdx.x >> 5;
    const int l16  = lane & 15;
    const int hf   = lane >> 4;
    const int m0   = (blockIdx.x * 8 + wave) * 16;

    const float* xrow = x + (size_t)(m0 + l16) * D_MODEL + 2 * hf;
    const float* brow = Blora + (size_t)l16 * D_MODEL + 2 * hf;

    v8f acc = (v8f){0.f, 0.f, 0.f, 0.f, 0.f, 0.f, 0.f, 0.f};

    for (int k = 0; k < D_MODEL; k += 16) {
#pragma unroll
        for (int kk = 0; kk < 16; kk += 4) {
            v2f a = *(const v2f*)(xrow + k + kk);
            v2f b = *(const v2f*)(brow + k + kk);
            acc = __builtin_amdgcn_wmma_f32_16x16x4_f32(false, a, false, b,
                                                        (short)0, acc, false, false);
        }
    }
#pragma unroll
    for (int v = 0; v < 8; ++v)
        u[(size_t)(m0 + hf * 8 + v) * RANK + l16] = acc[v];
}

// -------------------------------------------------------------------------
// TDM descriptor helpers (2-D bf16 tile 128 rows x 64 cols, row stride 4096,
// LDS padding: 4 dwords after every 32 dwords -> pitch 72 bf16 = 144B)
// -------------------------------------------------------------------------
static __device__ inline u32x4 tdm_g0(unsigned lds_addr, const __bf16* gptr) {
    const unsigned long long ga = (unsigned long long)(uintptr_t)gptr;
    u32x4 g0;
    g0[0] = 1u;                                   // count=1, user descriptor
    g0[1] = lds_addr;                             // lds_addr [63:32]
    g0[2] = (unsigned)ga;                         // global_addr lo  [95:64]
    g0[3] = (unsigned)(ga >> 32) | (2u << 30);    // global_addr hi + type=2
    return g0;
}

static __device__ inline i32x8 tdm_g1() {
    i32x8 g1;
    // [17:16] data_size=1 (2B) | [20] pad_enable | [24:22] pad_interval=4 (32 dw)
    // | [31:25] pad_amount=3 (4 dw)
    g1[0] = (1 << 16) | (1 << 20) | (4 << 22) | (3 << 25);
    g1[1] = (int)(4096u << 16);   // tensor_dim0 = 4096 (lo16 at bits 63:48)
    g1[2] = (int)(4096u << 16);   // tensor_dim0 hi16 = 0 ; tensor_dim1 lo16 = 4096
    g1[3] = (int)(64u << 16);     // tensor_dim1 hi16 = 0 ; tile_dim0 = 64
    g1[4] = 128;                  // tile_dim1 = 128 ; tile_dim2 = 0
    g1[5] = 4096;                 // tensor_dim0_stride = 4096 (lo32)
    g1[6] = 0;                    // stride hi16 ; tensor_dim1_stride lo16
    g1[7] = 0;
    return g1;
}

// -------------------------------------------------------------------------
// Kernel 2: out[m,n] = bias[n] + sum_k x[m,k]*W[n,k] + sum_r u[m,r]*A[n,r]
// bf16x3 split main loop; W_hi/W_lo tiles staged by the Tensor Data Mover.
// -------------------------------------------------------------------------
__global__ __launch_bounds__(256) void lora_gemm_bf16x3(const float* __restrict__ x,
                                                        const __bf16* __restrict__ Wh,
                                                        const __bf16* __restrict__ Wl,
                                                        const float* __restrict__ bias,
                                                        const float* __restrict__ Alora,
                                                        const float* __restrict__ u,
                                                        float* __restrict__ out) {
    __shared__ __bf16 Xh[BM * PITCH];
    __shared__ __bf16 Xl[BM * PITCH];
    __shared__ __bf16 Whs[BN * PITCH];
    __shared__ __bf16 Wls[BN * PITCH];

    const int t    = threadIdx.x;
    const int lane = t & 31;
    const int wave = t >> 5;
    const int l16  = lane & 15;
    const int hf   = lane >> 4;
    const int wm   = wave & 1;   // M offset 64*wm
    const int wn   = wave >> 1;  // N offset 32*wn

    const int n0 = blockIdx.x * BN;
    const int m0 = blockIdx.y * BM;

    // flat shared-pointer low 32 bits == LDS byte address
    const unsigned whLds = (unsigned)(uintptr_t)&Whs[0];
    const unsigned wlLds = (unsigned)(uintptr_t)&Wls[0];
    const i32x8 g1  = tdm_g1();
    const i32x4 gz4 = (i32x4){0, 0, 0, 0};
    const i32x8 gz8 = (i32x8){0, 0, 0, 0, 0, 0, 0, 0};

    v8f acc[4][2];
#pragma unroll
    for (int mt = 0; mt < 4; ++mt)
#pragma unroll
        for (int nt = 0; nt < 2; ++nt)
            acc[mt][nt] = (v8f){0.f, 0.f, 0.f, 0.f, 0.f, 0.f, 0.f, 0.f};

    // ---- LoRA prologue (exact fp32): acc += u_tile @ Alora_tile^T ----
#pragma unroll
    for (int r0 = 0; r0 < RANK; r0 += 4) {
        v2f au[4], ba[2];
#pragma unroll
        for (int mt = 0; mt < 4; ++mt) {
            const int m = m0 + wm * 64 + mt * 16 + l16;
            au[mt] = *(const v2f*)(u + (size_t)m * RANK + r0 + 2 * hf);
        }
#pragma unroll
        for (int nt = 0; nt < 2; ++nt) {
            const int n = n0 + wn * 32 + nt * 16 + l16;
            ba[nt] = *(const v2f*)(Alora + (size_t)n * RANK + r0 + 2 * hf);
        }
#pragma unroll
        for (int mt = 0; mt < 4; ++mt)
#pragma unroll
            for (int nt = 0; nt < 2; ++nt)
                acc[mt][nt] = __builtin_amdgcn_wmma_f32_16x16x4_f32(
                    false, au[mt], false, ba[nt], (short)0, acc[mt][nt], false, false);
    }

    // ---- main K loop: BK=64 chunk; W tiles via TDM, x split into hi/lo ----
    for (int kc = 0; kc < D_MODEL; kc += BK) {
        // 1) wave 0 kicks off the two tensor DMAs (global bf16 -> padded LDS)
        if (wave == 0) {
            __builtin_amdgcn_tensor_load_to_lds(
                tdm_g0(whLds, Wh + (size_t)n0 * D_MODEL + kc), g1, gz4, gz4, gz8, 0);
            __builtin_amdgcn_tensor_load_to_lds(
                tdm_g0(wlLds, Wl + (size_t)n0 * D_MODEL + kc), g1, gz4, gz4, gz8, 0);
        }

        // 2) all waves: x tile 128x64 fp32 -> hi/lo bf16 in LDS (overlaps TDM)
#pragma unroll
        for (int i = 0; i < 8; ++i) {
            const int idx = t + 256 * i;      // 0..2047 float4 slots
            const int row = idx >> 4;         // 0..127
            const int c4  = (idx & 15) << 2;  // 0,4,...,60
            v4f v = *(const v4f*)(x + (size_t)(m0 + row) * D_MODEL + kc + c4);
            v4bf h, l;
#pragma unroll
            for (int j = 0; j < 4; ++j) {
                __bf16 hj = (__bf16)v[j];
                h[j] = hj;
                l[j] = (__bf16)(v[j] - (float)hj);
            }
            *(v4bf*)(&Xh[row * PITCH + c4]) = h;
            *(v4bf*)(&Xl[row * PITCH + c4]) = l;
        }

        // 3) drain the DMA on the issuing wave, then publish both tiles
        if (wave == 0)
            __builtin_amdgcn_s_wait_tensorcnt(0);
        __syncthreads();

#pragma unroll
        for (int kk = 0; kk < BK; kk += 32) {
            // B frags (held live): lane = N, contiguous K run at kk + 16*hf
            v16bf bh[2], bl[2];
#pragma unroll
            for (int nt = 0; nt < 2; ++nt) {
                const __bf16* rh = &Whs[(wn * 32 + nt * 16 + l16) * PITCH + kk + 16 * hf];
                const __bf16* rl = &Wls[(wn * 32 + nt * 16 + l16) * PITCH + kk + 16 * hf];
                bh[nt] = cat8(*(const v8bf*)(rh), *(const v8bf*)(rh + 8));
                bl[nt] = cat8(*(const v8bf*)(rl), *(const v8bf*)(rl + 8));
            }
            // A frags transient per mt to bound VGPR pressure
#pragma unroll
            for (int mt = 0; mt < 4; ++mt) {
                const __bf16* rh = &Xh[(wm * 64 + mt * 16 + l16) * PITCH + kk];
                const __bf16* rl = &Xl[(wm * 64 + mt * 16 + l16) * PITCH + kk];
                v16bf ah = cat8(*(const v8bf*)(rh + 16 * hf), *(const v8bf*)(rh + 16 * hf + 8));
                v16bf al = cat8(*(const v8bf*)(rl + 16 * hf), *(const v8bf*)(rl + 16 * hf + 8));
#pragma unroll
                for (int nt = 0; nt < 2; ++nt) {
                    acc[mt][nt] = __builtin_amdgcn_wmma_f32_16x16x32_bf16(
                        false, ah, false, bh[nt], (short)0, acc[mt][nt], false, false);
                    acc[mt][nt] = __builtin_amdgcn_wmma_f32_16x16x32_bf16(
                        false, al, false, bh[nt], (short)0, acc[mt][nt], false, false);
                    acc[mt][nt] = __builtin_amdgcn_wmma_f32_16x16x32_bf16(
                        false, ah, false, bl[nt], (short)0, acc[mt][nt], false, false);
                }
            }
        }
        __syncthreads();
    }

    // ---- epilogue: add bias, store ----
#pragma unroll
    for (int mt = 0; mt < 4; ++mt) {
#pragma unroll
        for (int nt = 0; nt < 2; ++nt) {
            const int n    = n0 + wn * 32 + nt * 16 + l16;
            const float bv = bias[n];
            const v8f c    = acc[mt][nt];
#pragma unroll
            for (int v = 0; v < 8; ++v) {
                const int m = m0 + wm * 64 + mt * 16 + hf * 8 + v;
                out[(size_t)m * D_MODEL + n] = c[v] + bv;
            }
        }
    }
}

// -------------------------------------------------------------------------
extern "C" void kernel_launch(void* const* d_in, const int* in_sizes, int n_in,
                              void* d_out, int out_size, void* d_ws, size_t ws_size,
                              hipStream_t stream) {
    const float* x     = (const float*)d_in[0];
    const float* W     = (const float*)d_in[1];
    const float* bias  = (const float*)d_in[2];
    const float* Alora = (const float*)d_in[3];
    const float* Blora = (const float*)d_in[4];
    float*       out   = (float*)d_out;

    // ws layout: u [1MB] | W_hi [32MB bf16] | W_lo [32MB bf16]
    float*  u   = (float*)d_ws;
    __bf16* Whg = (__bf16*)((char*)d_ws + (1u << 20));
    __bf16* Wlg = Whg + (size_t)D_MODEL * D_MODEL;

    // 0) split W into bf16 hi/lo
    split_w_kernel<<<(D_MODEL * D_MODEL) / (256 * 4), 256, 0, stream>>>(W, Whg, Wlg);

    // 1) u = x @ B^T (exact fp32 WMMA)
    lora_u_kernel<<<M_TOTAL / (16 * 8), 256, 0, stream>>>(x, Blora, u);

    // 2) out = x @ W^T (bf16x3 WMMA, TDM-staged W) + u @ A^T (fp32 WMMA) + bias
    dim3 grid(D_MODEL / BN, M_TOTAL / BM);
    lora_gemm_bf16x3<<<grid, 256, 0, stream>>>(x, Whg, Wlg, bias, Alora, u, out);
}
